// Mel2Audio_14413910245529
// MI455X (gfx1250) — matrix operand order repaired and checked by reading, compiled, not run on hardware
//
#include <hip/hip_runtime.h>

// ---------------- problem constants ----------------
#define NFFT     1024
#define HOP      64
#define NFRAMES  512
#define NBINS    513
#define SIGLEN   33728
#define NMELS    80
#define NITER    300

// padded GEMM dims
#define KI       1088        // istft K: 2 * 544 (Re | Im, bins padded 513->544)
#define KI_STEPS 34          // 1088 / 32
#define MI_TILES 64          // 1024 / 16 time samples
#define MF_TILES 66          // stft M: 2*528 rows (Re 0..527 | Im 528..1055)
#define MP_TILES 33          // paired Re/Im tiles (528/16)
#define KF_STEPS 32          // stft K = 1024 samples / 32
#define NT       32          // 512 frames / 16

typedef __attribute__((ext_vector_type(16))) __bf16 v16bf;
typedef __attribute__((ext_vector_type(8)))  float  v8f;

__device__ __forceinline__ unsigned short f32_bf16(float f) {
  unsigned int u = __float_as_uint(f);
  u += 0x7FFFu + ((u >> 16) & 1u);          // round to nearest even
  return (unsigned short)(u >> 16);
}

__device__ __forceinline__ float hann(int n) {
  return 0.5f - 0.5f * __cosf(6.28318530717958647692f * (float)n * (1.0f / 1024.0f));
}

// A-operand lane/element -> K mapping (16-bit A 16x32 tile)
__device__ __forceinline__ int kA_of(int e, int lane) {
  return e + (((lane >> 4) & 1) << 4);
}
// B-operand lane/element -> K mapping (16-bit B 32x16 tile)
__device__ __forceinline__ int kB_of(int e, int lane) {
  return e + (((lane >> 4) & 1) << 4);
}

// ---------------- WMMA GEMM: C[M x N] = A @ B, bf16 in / f32 out -------------
// A pre-swizzled: idx = ((mT*kSteps + ks)*32 + lane)*16 + e
// B pre-swizzled: idx = ((ks*nTiles + nT)*32 + lane)*16 + e
// one wave -> one 16x16 C tile; grid exact so EXEC is all ones.
__global__ void k_gemm_bf16(const unsigned short* __restrict__ A,
                            const unsigned short* __restrict__ B,
                            float* __restrict__ C,
                            int nTiles, int kSteps, int ldc) {
  int wave = threadIdx.x >> 5;
  int lane = threadIdx.x & 31;
  int tile = blockIdx.x * 4 + wave;
  int mT = tile / nTiles;
  int nT = tile % nTiles;
  const unsigned short* ap = A + ((size_t)(mT * kSteps) * 32 + lane) * 16;
  const unsigned short* bp = B + ((size_t)nT * 32 + lane) * 16;
  size_t bstride = (size_t)nTiles * 512;
  v8f acc = {};
  for (int ks = 0; ks < kSteps; ++ks) {
    v16bf a = *(const v16bf*)ap;
    v16bf b = *(const v16bf*)bp;
    acc = __builtin_amdgcn_wmma_f32_16x16x32_bf16(false, a, false, b,
                                                  (short)0, acc, false, false);
    ap += 512;
    bp += bstride;
  }
  int col = nT * 16 + (lane & 15);
  int row = mT * 16 + ((lane >> 4) << 3);
  float* cp = C + (size_t)row * ldc + col;
#pragma unroll
  for (int r = 0; r < 8; ++r) cp[(size_t)r * ldc] = acc[r];
}

// ---- stft GEMM with paired Re/Im tiles + fused Griffin-Lim phase epilogue ---
// Wave computes Re tile mT and Im tile mT+33 over the same B fragment
// (2 WMMAs per K-step, B loaded once), then writes X = mag*est/|est|
// directly into the pre-swizzled istft B operand XB.
__global__ void k_stft_phase(const unsigned short* __restrict__ AF,
                             const unsigned short* __restrict__ B,
                             const float* __restrict__ mag,
                             unsigned short* __restrict__ XB) {
  int wave = threadIdx.x >> 5;
  int lane = threadIdx.x & 31;
  int tile = blockIdx.x * 4 + wave;   // 0..1055
  int mT = tile / NT;                 // 0..32 (Re tile; Im tile = mT+33)
  int nT = tile % NT;
  const unsigned short* apR = AF + ((size_t)(mT * KF_STEPS) * 32 + lane) * 16;
  const unsigned short* apI = AF + ((size_t)((mT + 33) * KF_STEPS) * 32 + lane) * 16;
  const unsigned short* bp  = B + ((size_t)nT * 32 + lane) * 16;
  size_t bstride = (size_t)NT * 512;
  v8f accR = {}, accI = {};
  for (int ks = 0; ks < KF_STEPS; ++ks) {
    v16bf b  = *(const v16bf*)bp;
    v16bf a0 = *(const v16bf*)apR;
    v16bf a1 = *(const v16bf*)apI;
    accR = __builtin_amdgcn_wmma_f32_16x16x32_bf16(false, a0, false, b,
                                                   (short)0, accR, false, false);
    accI = __builtin_amdgcn_wmma_f32_16x16x32_bf16(false, a1, false, b,
                                                   (short)0, accI, false, false);
    apR += 512; apI += 512; bp += bstride;
  }
  int t     = nT * 16 + (lane & 15);
  int rbase = mT * 16 + ((lane >> 4) << 3);
#pragma unroll
  for (int r = 0; r < 8; ++r) {
    int bin = rbase + r;
    if (bin > 512) continue;                  // pad rows: XB pads stay zero
    float re = accR[r], im = accI[r];
    float rad = __fsqrt_rn(re * re + im * im);
    if (rad < 1e-8f) rad = 1e-8f;
    float sc = mag[bin * NFRAMES + t] / rad;
    int kkR = bin, kkI = bin + 544;           // Re slot | Im slot in K=1088
    int klR = kkR & 31, klI = kkI & 31;
    size_t idxR = ((size_t)((kkR >> 5) * NT + nT) * 32 +
                   ((klR >> 4) << 4) + (lane & 15)) * 16 + (klR & 15);
    size_t idxI = ((size_t)((kkI >> 5) * NT + nT) * 32 +
                   ((klI >> 4) << 4) + (lane & 15)) * 16 + (klI & 15);
    XB[idxR] = f32_bf16(re * sc);
    XB[idxI] = f32_bf16(im * sc);
  }
}

// ---------------- setup: inverse-DFT matrix (istft), window folded in -------
__global__ void k_setup_AI(unsigned short* __restrict__ AI) {
  int idx = blockIdx.x * blockDim.x + threadIdx.x;
  const int total = MI_TILES * KI_STEPS * 512;
  if (idx >= total) return;
  int e = idx & 15, lane = (idx >> 4) & 31, ts = idx >> 9;
  int ks = ts % KI_STEPS, mT = ts / KI_STEPS;
  int n  = mT * 16 + (lane & 15);
  int kk = ks * 32 + kA_of(e, lane);
  float val = 0.0f;
  int k  = (kk < 544) ? kk : kk - 544;
  int im = (kk >= 544);
  if (k <= 512) {
    float w  = (k == 0 || k == 512) ? 1.0f : 2.0f;
    float th = 6.28318530717958647692f *
               (float)((k * n) & 1023) * (1.0f / 1024.0f);
    float tr = im ? -__sinf(th) : __cosf(th);
    val = w * tr * (1.0f / 1024.0f) * hann(n);
  }
  AI[idx] = f32_bf16(val);
}

// ---------------- setup: forward-DFT matrix (stft), window folded in --------
__global__ void k_setup_AF(unsigned short* __restrict__ AF) {
  int idx = blockIdx.x * blockDim.x + threadIdx.x;
  const int total = MF_TILES * KF_STEPS * 512;
  if (idx >= total) return;
  int e = idx & 15, lane = (idx >> 4) & 31, ts = idx >> 9;
  int ks = ts % KF_STEPS, mT = ts / KF_STEPS;
  int m = mT * 16 + (lane & 15);
  int n = ks * 32 + kA_of(e, lane);
  float val = 0.0f;
  int k  = (m < 528) ? m : m - 528;
  int im = (m >= 528);
  if (k <= 512) {
    float th = 6.28318530717958647692f *
               (float)((k * n) & 1023) * (1.0f / 1024.0f);
    val = (im ? -__sinf(th) : __cosf(th)) * hann(n);
  }
  AF[idx] = f32_bf16(val);
}

// ---------------- setup: 1 / window-sum-squares ------------------------------
__global__ void k_setup_wssinv(float* __restrict__ wi) {
  int s = blockIdx.x * blockDim.x + threadIdx.x;
  if (s >= SIGLEN) return;
  int tmin = s - 1023; tmin = (tmin > 0) ? ((tmin + 63) >> 6) : 0;
  int tmax = s >> 6;   if (tmax > NFRAMES - 1) tmax = NFRAMES - 1;
  float acc = 0.0f;
  for (int t = tmin; t <= tmax; ++t) {
    float w = hann(s - (t << 6));
    acc += w * w;
  }
  wi[s] = (acc > 1e-11f) ? (1.0f / acc) : 1.0f;
}

// ---------------- mag = pinv(mel_basis) @ 10^mel (full f32) ------------------
__global__ void k_pow10(const float* __restrict__ mel, float* __restrict__ p10) {
  int i = blockIdx.x * blockDim.x + threadIdx.x;
  if (i >= NMELS * NFRAMES) return;
  p10[i] = exp2f(mel[i] * 3.32192809488736234787f);   // 10^x = 2^(x*log2 10)
}
__global__ void k_mag(const float* __restrict__ invmel,
                      const float* __restrict__ p10,
                      float* __restrict__ mag) {
  int idx = blockIdx.x * blockDim.x + threadIdx.x;
  if (idx >= NBINS * NFRAMES) return;
  int t = idx & 511, i = idx >> 9;
  float acc = 0.0f;
  for (int j = 0; j < NMELS; ++j)
    acc = fmaf(invmel[i * NMELS + j], p10[j * NFRAMES + t], acc);
  mag[idx] = acc;
}

// ---------------- X0 = mag (real), pre-swizzled bf16 B operand ---------------
// Also zeroes the pad K-slots, which later iterations never touch.
__global__ void k_initXB(const float* __restrict__ mag,
                         unsigned short* __restrict__ XB) {
  int idx = blockIdx.x * blockDim.x + threadIdx.x;
  const int total = KI_STEPS * NT * 512;
  if (idx >= total) return;
  int e = idx & 15, lane = (idx >> 4) & 31, ts = idx >> 9;
  int nT = ts % NT, ks = ts / NT;
  int t  = nT * 16 + (lane & 15);
  int kk = ks * 32 + kB_of(e, lane);
  float v = 0.0f;
  if (kk < 544 && kk <= 512) v = mag[kk * NFRAMES + t];
  XB[idx] = f32_bf16(v);
}

// -------- fused overlap-add + NOLA normalize + frame into stft B operand -----
__global__ void k_olaframe(const float* __restrict__ frames,
                           const float* __restrict__ wssinv,
                           unsigned short* __restrict__ FR) {
  int idx = blockIdx.x * blockDim.x + threadIdx.x;
  const int total = KF_STEPS * NT * 512;
  if (idx >= total) return;
  int e = idx & 15, lane = (idx >> 4) & 31, ts = idx >> 9;
  int nT = ts % NT, ks = ts / NT;
  int t  = nT * 16 + (lane & 15);
  int kk = ks * 32 + kB_of(e, lane);
  int s  = t * HOP + kk;
  int tmin = s - 1023; tmin = (tmin > 0) ? ((tmin + 63) >> 6) : 0;
  int tmax = s >> 6;   if (tmax > NFRAMES - 1) tmax = NFRAMES - 1;
  float acc = 0.0f;
  for (int tt = tmin; tt <= tmax; ++tt)
    acc += frames[(size_t)(s - (tt << 6)) * NFRAMES + tt];
  FR[idx] = f32_bf16(acc * wssinv[s]);
}

// ---------------- final overlap-add -> output signal -------------------------
__global__ void k_ola(const float* __restrict__ frames,
                      const float* __restrict__ wssinv,
                      float* __restrict__ out) {
  int s = blockIdx.x * blockDim.x + threadIdx.x;
  if (s >= SIGLEN) return;
  int tmin = s - 1023; tmin = (tmin > 0) ? ((tmin + 63) >> 6) : 0;
  int tmax = s >> 6;   if (tmax > NFRAMES - 1) tmax = NFRAMES - 1;
  float acc = 0.0f;
  for (int t = tmin; t <= tmax; ++t)
    acc += frames[(size_t)(s - (t << 6)) * NFRAMES + t];
  out[s] = acc * wssinv[s];
}

// ---------------- host driver ------------------------------------------------
extern "C" void kernel_launch(void* const* d_in, const int* in_sizes, int n_in,
                              void* d_out, int out_size, void* d_ws, size_t ws_size,
                              hipStream_t stream) {
  (void)in_sizes; (void)n_in; (void)out_size; (void)ws_size;
  const float* mel    = (const float*)d_in[0];   // [80, 512]
  const float* invmel = (const float*)d_in[1];   // [513, 80]
  float* out = (float*)d_out;                    // [33728]

  char* base = (char*)d_ws;
  size_t off = 0;
  auto alloc = [&](size_t bytes) -> void* {
    void* p = base + off;
    off += (bytes + 255) & ~(size_t)255;
    return p;
  };
  unsigned short* AI  = (unsigned short*)alloc((size_t)MI_TILES * KI_STEPS * 512 * 2);
  unsigned short* AF  = (unsigned short*)alloc((size_t)MF_TILES * KF_STEPS * 512 * 2);
  unsigned short* XB  = (unsigned short*)alloc((size_t)KI_STEPS * NT * 512 * 2);
  unsigned short* FR  = (unsigned short*)alloc((size_t)KF_STEPS * NT * 512 * 2);
  float* WSSI = (float*)alloc((size_t)SIGLEN * 4);
  float* P10  = (float*)alloc((size_t)NMELS * NFRAMES * 4);
  float* MAG  = (float*)alloc((size_t)NBINS * NFRAMES * 4);
  float* FRM  = (float*)alloc((size_t)NFFT * NFRAMES * 4);   // istft GEMM out

  const int TPB = 256;
  auto blocks = [](int n) { return (n + 255) / 256; };

  // per-launch setup — deterministic, graph-capturable
  k_setup_AI    <<<blocks(MI_TILES * KI_STEPS * 512), TPB, 0, stream>>>(AI);
  k_setup_AF    <<<blocks(MF_TILES * KF_STEPS * 512), TPB, 0, stream>>>(AF);
  k_setup_wssinv<<<blocks(SIGLEN), TPB, 0, stream>>>(WSSI);
  k_pow10       <<<blocks(NMELS * NFRAMES), TPB, 0, stream>>>(mel, P10);
  k_mag         <<<blocks(NBINS * NFRAMES), TPB, 0, stream>>>(invmel, P10, MAG);
  k_initXB      <<<blocks(KI_STEPS * NT * 512), TPB, 0, stream>>>(MAG, XB);

  const int GI = MI_TILES * NT / 4;   // 512 blocks (istft GEMM)
  const int GP = MP_TILES * NT / 4;   // 264 blocks (paired stft GEMM + phase)

  for (int it = 0; it < NITER; ++it) {
    k_gemm_bf16 <<<GI, 128, 0, stream>>>(AI, XB, FRM, NT, KI_STEPS, NFRAMES);
    k_olaframe  <<<blocks(KF_STEPS * NT * 512), TPB, 0, stream>>>(FRM, WSSI, FR);
    k_stft_phase<<<GP, 128, 0, stream>>>(AF, FR, MAG, XB);
  }

  // final istft -> output signal
  k_gemm_bf16<<<GI, 128, 0, stream>>>(AI, XB, FRM, NT, KI_STEPS, NFRAMES);
  k_ola      <<<blocks(SIGLEN), TPB, 0, stream>>>(FRM, WSSI, out);
}
